// GCN_Encoder_4604204941836
// MI455X (gfx1250) — matrix operand order repaired and checked by reading, not compiled, not run on hardware
//
#include <hip/hip_runtime.h>
#include <hip/hip_bf16.h>
#include <stdint.h>

// ---------------------------------------------------------------------------
// GCN encoder for MI455X (gfx1250, wave32).
// GEMMs via v_wmma_f32_16x16x32_bf16 (bf16 A/B, fp32 accum); everything else
// (degree norm, edge scatter-add, BN, pooling) in fp32 VALU + L2 atomics.
// ---------------------------------------------------------------------------

typedef unsigned short ushort_t;
typedef __attribute__((ext_vector_type(16))) __bf16 v16bf;
typedef __attribute__((ext_vector_type(8)))  float  v8f;

#define NEG_SLOPE 0.01f
#define BN_EPS    1e-5f
#define CDIV(a,b) (((a)+(b)-1)/(b))

__device__ __forceinline__ ushort_t f2bf(float f) {
  union { float f; unsigned u; } a; a.f = f;
  unsigned u = a.u;
  unsigned lsb = (u >> 16) & 1u;          // round-to-nearest-even
  u += 0x7FFFu + lsb;
  return (ushort_t)(u >> 16);
}

union Frag16 { v16bf v; uint4 q[2]; };

// ---------------------------------------------------------------------------
// Small utility kernels
// ---------------------------------------------------------------------------
__global__ void k_fill0(float* p, int n) {
  int i = blockIdx.x * blockDim.x + threadIdx.x;
  if (i < n) p[i] = 0.f;
}

__global__ void k_deg(const long long* dst, float* deg, int E) {
  int e = blockIdx.x * blockDim.x + threadIdx.x;
  if (e < E) atomicAdd(&deg[(int)dst[e]], 1.0f);
}

__global__ void k_degfin(float* deg, float* dinv, float* selfn, int N) {
  int i = blockIdx.x * blockDim.x + threadIdx.x;
  if (i < N) {
    float d = deg[i] + 1.0f;               // +1 for self loop
    dinv[i]  = rsqrtf(d);
    selfn[i] = 1.0f / d;
  }
}

__global__ void k_edgenorm(const long long* src, const long long* dst,
                           const float* dinv, float* norm, int E) {
  int e = blockIdx.x * blockDim.x + threadIdx.x;
  if (e < E) norm[e] = dinv[(int)src[e]] * dinv[(int)dst[e]];
}

__global__ void k_f2bf(const float* x, ushort_t* y, int n) {
  int i = blockIdx.x * blockDim.x + threadIdx.x;
  if (i < n) y[i] = f2bf(x[i]);
}

// W[di][do] (row-major) -> Wt[do][di] bf16 (so B-fragments are contiguous)
__global__ void k_wt(const float* W, ushort_t* Wt, int di, int dof) {
  int i = blockIdx.x * blockDim.x + threadIdx.x;
  if (i < di * dof) {
    int k = i / dof, c = i - k * dof;
    Wt[(size_t)c * di + k] = f2bf(W[(size_t)k * dof + c]);
  }
}

// ---------------------------------------------------------------------------
// WMMA GEMM: T[N x dout] = Xbf[N x K] @ W  (W given transposed-bf16 Wt[dout x K])
// Each wave computes a 16x64 strip: 1 A-fragment reused by 4 WMMAs per K-step.
// Block = 256 threads = 8 waves -> 128 rows x 64 cols per block.
// K is a template param so the K-loop fully unrolls and the backend can
// hoist/interleave the b128 fragment loads across WMMAs (staggered waits).
// ---------------------------------------------------------------------------
template <int K>
__global__ void __launch_bounds__(256)
k_gemm(const ushort_t* __restrict__ X, const ushort_t* __restrict__ Wt,
       float* __restrict__ T, int Nrows, int dout) {
  const int lane = threadIdx.x & 31;
  const int wave = threadIdx.x >> 5;
  const int rowBase = blockIdx.x * 128 + wave * 16;
  const int colBase = blockIdx.y * 64;
  const int l15   = lane & 15;
  const int kbase = (lane >> 4) * 8;     // A: lanes 16-31 start at K+8
  const int khalf = (lane >> 4) * 8;     // B: lanes 16-31 start at K+8

  v8f acc[4] = {};

  int arow = rowBase + l15;
  if (arow >= Nrows) arow = Nrows - 1;   // clamp (stores are guarded)
  const ushort_t* aptr = X + (size_t)arow * K + kbase;

  const ushort_t* bcol[4];
#pragma unroll
  for (int j = 0; j < 4; ++j)
    bcol[j] = Wt + (size_t)(colBase + j * 16 + l15) * K + khalf;

#pragma unroll
  for (int k0 = 0; k0 < K; k0 += 32) {
    Frag16 fa;
    fa.q[0] = *reinterpret_cast<const uint4*>(aptr + k0);        // K {0..7}+kbase
    fa.q[1] = *reinterpret_cast<const uint4*>(aptr + k0 + 16);   // K {16..23}+kbase
    Frag16 fb[4];
#pragma unroll
    for (int j = 0; j < 4; ++j) {                                // distinct regs
      fb[j].q[0] = *reinterpret_cast<const uint4*>(bcol[j] + k0);
      fb[j].q[1] = *reinterpret_cast<const uint4*>(bcol[j] + k0 + 16);
    }
#pragma unroll
    for (int j = 0; j < 4; ++j)
      acc[j] = __builtin_amdgcn_wmma_f32_16x16x32_bf16(
          false, fa.v, false, fb[j].v, (short)0, acc[j], false, false);
  }

  const int mrow0 = rowBase + (lane >> 4) * 8;   // C/D: M = r + 8*(lane>=16)
#pragma unroll
  for (int j = 0; j < 4; ++j) {
    int col = colBase + j * 16 + l15;            // N = lane & 15
#pragma unroll
    for (int r = 0; r < 8; ++r) {
      int row = mrow0 + r;
      if (row < Nrows) T[(size_t)row * dout + col] = acc[j][r];
    }
  }
}

// ---------------------------------------------------------------------------
// Aggregation: H = self_norm * T + b  (init) ; then edge scatter-add
// ---------------------------------------------------------------------------
__global__ void k_agginit(const float* T, const float* selfn, const float* bias,
                          float* H, int total, int dout) {
  int i = blockIdx.x * blockDim.x + threadIdx.x;
  if (i >= total) return;
  int sh = 31 - __clz(dout);               // dout is 128 or 256
  int n = i >> sh, c = i & (dout - 1);
  H[i] = T[i] * selfn[n] + bias[c];
}

// One thread = (edge, 4 channels): amortize index/norm loads, float4 gather,
// 4 L2-resident global_atomic_add_f32 per thread.
__global__ void k_scatter4(const long long* __restrict__ src,
                           const long long* __restrict__ dst,
                           const float* __restrict__ norm,
                           const float* __restrict__ T,
                           float* __restrict__ H, int total4, int dout) {
  int i = blockIdx.x * blockDim.x + threadIdx.x;
  if (i >= total4) return;
  int q  = dout >> 2;                      // channel-quads per edge
  int sh = 31 - __clz(q);
  int e  = i >> sh;
  int c0 = (i & (q - 1)) << 2;
  int s = (int)src[e], d = (int)dst[e];
  float w = norm[e];
  float4 t = *reinterpret_cast<const float4*>(T + (size_t)s * dout + c0);
  float* hp = H + (size_t)d * dout + c0;
  atomicAdd(hp + 0, t.x * w);
  atomicAdd(hp + 1, t.y * w);
  atomicAdd(hp + 2, t.z * w);
  atomicAdd(hp + 3, t.w * w);
}

// ---------------------------------------------------------------------------
// BatchNorm (training stats) over LeakyReLU(H)
// ---------------------------------------------------------------------------
__global__ void k_bnstats(const float* H, float* bsum, float* bsqs,
                          int N, int dout) {
  int c  = threadIdx.x;                    // blockDim.x == dout
  int r0 = blockIdx.x * 128;
  int r1 = r0 + 128; if (r1 > N) r1 = N;
  float s = 0.f, s2 = 0.f;
  for (int r = r0; r < r1; ++r) {
    float x = H[(size_t)r * dout + c];
    float y = x > 0.f ? x : NEG_SLOPE * x;
    s += y; s2 += y * y;
  }
  atomicAdd(&bsum[c], s);
  atomicAdd(&bsqs[c], s2);
}

__global__ void k_bnfin(const float* bsum, const float* bsqs,
                        const float* g, const float* bt,
                        float* scale, float* shift, int N, int dout) {
  int c = threadIdx.x;
  if (c >= dout) return;
  float inv = 1.0f / (float)N;
  float mu  = bsum[c] * inv;
  float var = fmaxf(bsqs[c] * inv - mu * mu, 0.f);
  float sc  = g[c] * rsqrtf(var + BN_EPS);
  scale[c] = sc;
  shift[c] = bt[c] - mu * sc;
}

// LeakyReLU + affine; emit fp32 (next agg / pooling) and bf16 (next GEMM)
__global__ void k_bnapply(const float* H, const float* scale, const float* shift,
                          ushort_t* Xbf, float* Tf, int total, int dout) {
  int i = blockIdx.x * blockDim.x + threadIdx.x;
  if (i >= total) return;
  int c = i & (dout - 1);
  float x = H[i];
  float y = x > 0.f ? x : NEG_SLOPE * x;
  float v = y * scale[c] + shift[c];
  Xbf[i] = f2bf(v);
  Tf[i]  = v;
}

// ---------------------------------------------------------------------------
// Global mean pool
// ---------------------------------------------------------------------------
__global__ void k_poolcnt(const long long* batch, float* gcnt, int N) {
  int n = blockIdx.x * blockDim.x + threadIdx.x;
  if (n < N) atomicAdd(&gcnt[(int)batch[n]], 1.0f);
}

__global__ void k_poolsum4(const long long* batch, const float* T,
                           float* gsum, int total4, int dout) {
  int i = blockIdx.x * blockDim.x + threadIdx.x;
  if (i >= total4) return;
  int q  = dout >> 2;
  int sh = 31 - __clz(q);
  int n  = i >> sh;
  int c0 = (i & (q - 1)) << 2;
  int b  = (int)batch[n];
  float4 t = *reinterpret_cast<const float4*>(T + (size_t)n * dout + c0);
  float* gp = gsum + (size_t)b * dout + c0;
  atomicAdd(gp + 0, t.x);
  atomicAdd(gp + 1, t.y);
  atomicAdd(gp + 2, t.z);
  atomicAdd(gp + 3, t.w);
}

__global__ void k_pooldiv(const float* gsum, const float* gcnt,
                          float* out, int total, int dout) {
  int i = blockIdx.x * blockDim.x + threadIdx.x;
  if (i >= total) return;
  int g = i / dout;
  out[i] = gsum[i] / fmaxf(gcnt[g], 1.0f);
}

// ---------------------------------------------------------------------------
// Host orchestration
// ---------------------------------------------------------------------------
extern "C" void kernel_launch(void* const* d_in, const int* in_sizes, int n_in,
                              void* d_out, int out_size, void* d_ws, size_t ws_size,
                              hipStream_t stream) {
  const float*     x     = (const float*)d_in[0];
  const long long* eidx  = (const long long*)d_in[1];   // int64 [2, E]
  const long long* batch = (const long long*)d_in[2];   // int64 [N]

  const int N = in_sizes[0] / 128;
  const int E = in_sizes[1] / 2;
  const int G = out_size / 128;

  const long long* esrc = eidx;
  const long long* edst = eidx + E;

  const float *W[4], *b[4], *g[4], *bt[4];
  for (int i = 0; i < 4; ++i) {
    W[i]  = (const float*)d_in[3 + 4 * i];
    b[i]  = (const float*)d_in[4 + 4 * i];
    g[i]  = (const float*)d_in[5 + 4 * i];
    bt[i] = (const float*)d_in[6 + 4 * i];
  }
  const int dims[5] = {128, 256, 256, 256, 128};

  // Workspace carve-up (256B aligned)
  char*  ws  = (char*)d_ws;
  size_t off = 0;
  auto alloc = [&](size_t bytes) -> void* {
    void* p = ws + off;
    off = (off + bytes + 255) & ~(size_t)255;
    return p;
  };
  float*    deg   = (float*)alloc((size_t)N * 4);
  float*    dinv  = (float*)alloc((size_t)N * 4);
  float*    selfn = (float*)alloc((size_t)N * 4);
  float*    norm  = (float*)alloc((size_t)E * 4);
  float*    T     = (float*)alloc((size_t)N * 256 * 4);
  float*    H     = (float*)alloc((size_t)N * 256 * 4);
  ushort_t* Xbf   = (ushort_t*)alloc((size_t)N * 256 * 2);
  ushort_t* Wt    = (ushort_t*)alloc(256 * 256 * 2);
  float*    bsum  = (float*)alloc(256 * 4);
  float*    bsqs  = (float*)alloc(256 * 4);
  float*    bscl  = (float*)alloc(256 * 4);
  float*    bsft  = (float*)alloc(256 * 4);
  float*    gsum  = (float*)alloc((size_t)G * 128 * 4);
  float*    gcnt  = (float*)alloc((size_t)G * 4);
  (void)ws_size; (void)n_in;

  // Degree / normalization (shared by all layers)
  k_fill0   <<<CDIV(N, 256), 256, 0, stream>>>(deg, N);
  k_deg     <<<CDIV(E, 256), 256, 0, stream>>>(edst, deg, E);
  k_degfin  <<<CDIV(N, 256), 256, 0, stream>>>(deg, dinv, selfn, N);
  k_edgenorm<<<CDIV(E, 256), 256, 0, stream>>>(esrc, edst, dinv, norm, E);

  // Input -> bf16 for first GEMM
  k_f2bf<<<CDIV(N * 128, 256), 256, 0, stream>>>(x, Xbf, N * 128);

  for (int L = 0; L < 4; ++L) {
    const int di = dims[L], dout = dims[L + 1];
    const int total = N * dout;

    k_wt<<<CDIV(di * dout, 256), 256, 0, stream>>>(W[L], Wt, di, dout);

    dim3 gg(CDIV(N, 128), dout / 64);
    if (di == 128)
      k_gemm<128><<<gg, 256, 0, stream>>>(Xbf, Wt, T, N, dout);
    else
      k_gemm<256><<<gg, 256, 0, stream>>>(Xbf, Wt, T, N, dout);

    k_agginit<<<CDIV(total, 256), 256, 0, stream>>>(T, selfn, b[L], H, total, dout);
    k_scatter4<<<CDIV(E * (dout / 4), 256), 256, 0, stream>>>(
        esrc, edst, norm, T, H, E * (dout / 4), dout);

    k_fill0<<<1, 256, 0, stream>>>(bsum, dout);
    k_fill0<<<1, 256, 0, stream>>>(bsqs, dout);
    k_bnstats<<<CDIV(N, 128), dout, 0, stream>>>(H, bsum, bsqs, N, dout);
    k_bnfin  <<<1, dout, 0, stream>>>(bsum, bsqs, g[L], bt[L], bscl, bsft, N, dout);
    k_bnapply<<<CDIV(total, 256), 256, 0, stream>>>(H, bscl, bsft, Xbf, T,
                                                    total, dout);
  }

  // Global mean pool over graphs
  k_fill0   <<<CDIV(G * 128, 256), 256, 0, stream>>>(gsum, G * 128);
  k_fill0   <<<1, 64, 0, stream>>>(gcnt, G);
  k_poolcnt <<<CDIV(N, 256), 256, 0, stream>>>(batch, gcnt, N);
  k_poolsum4<<<CDIV(N * 32, 256), 256, 0, stream>>>(batch, T, gsum, N * 32, 128);
  k_pooldiv <<<CDIV(G * 128, 256), 256, 0, stream>>>(gsum, gcnt, (float*)d_out,
                                                     G, 128);
}
